// GriffinHaloBase_76459007803766
// MI455X (gfx1250) — matrix operand order repaired and checked by reading, not compiled, not run on hardware
//
#include <hip/hip_runtime.h>

// ---------------------------------------------------------------------------
// Griffin block for MI455X (gfx1250): bf16 WMMA GEMMs (pipelined, 32x64/wave)
// + chunked-parallel RG-LRU scan. Workspace (bytes from d_ws), peak ~171 MB:
//   [  0MB) WT_PROJ  bf16 [1920 x 768]   (conv_in | v | r | i, transposed)
//   [  4MB) OUT_WT   bf16 [ 768 x 768]
//   [  8MB) W1T      bf16 [2048 x 768]
//   [ 12MB) W3T      bf16 [2048 x 768]
//   [ 16MB) W2T      bf16 [ 768 x 2048]
//   [ 20MB) XN       bf16 [8192 x 768]   (also reused as NORMED2)
//   [ 36MB) PROJ     f32  [8192 x 1920]  (also reused as H bf16 [8192x2048])
//   [100MB) MIXCAT   bf16 [8192 x 768]
//   [116MB) MIXOUT   f32  [8192 x 768]   (also reused as FFNOUT)
//   [144MB) XMID     f32  [8192 x 768]
//   [170MB) SA/SH/HST f32 3 x [4*384*16] (scan chunk states)
// ---------------------------------------------------------------------------

typedef __bf16 bf16_t;
typedef __attribute__((ext_vector_type(16))) __bf16 v16bf;
typedef __attribute__((ext_vector_type(8)))  __bf16 v8bf;
typedef __attribute__((ext_vector_type(8)))  float  v8f;

#define NB   4
#define TT   2048
#define DD   768
#define DC   384
#define DG   384
#define DF   2048
#define NTOK (NB * TT)     // 8192
#define NDE  (NTOK * DD)   // 6291456
#define CHUNK 128
#define NCH   (TT / CHUNK) // 16

__device__ __forceinline__ bf16_t f2bf(float f) {
    unsigned u = __builtin_bit_cast(unsigned, f);
    unsigned r = u + 0x7FFFu + ((u >> 16) & 1u);   // round-to-nearest-even
    unsigned short s = (unsigned short)(r >> 16);
    return __builtin_bit_cast(bf16_t, s);
}

__device__ __forceinline__ float sigmoidf_(float x) {
    return 1.0f / (1.0f + __expf(-x));
}

__device__ __forceinline__ v8f wmma_bf16(v16bf a, v16bf b, v8f c) {
    return __builtin_amdgcn_wmma_f32_16x16x32_bf16(
        false, a, false, b, (short)0, c, false, false);
}

// A-fragment loader (ISA 7.12.2 16-bit A 16x32 layout): lane holds row M=lane&15,
// K = {kb..kb+7, kb+16..kb+23}, kb = 8*(lane>=16). base is pre-offset by kb.
__device__ __forceinline__ v16bf load_afrag(const bf16_t* base, int k0) {
    v8bf lo = *(const v8bf*)(base + k0);
    v8bf hi = *(const v8bf*)(base + k0 + 16);
    v16bf a;
#pragma unroll
    for (int e = 0; e < 8; ++e) { a[e] = lo[e]; a[8 + e] = hi[e]; }
    return a;
}

// ---------------- weight transpose-convert: f32 [K,Nc] -> bf16 [Nc,K] -------
__global__ void convertT_k(const float* __restrict__ in, int K, int Nc,
                           bf16_t* __restrict__ out) {
    int idx = blockIdx.x * blockDim.x + threadIdx.x;
    if (idx >= K * Nc) return;
    int k = idx / Nc;
    int c = idx - k * Nc;
    out[(size_t)c * K + k] = f2bf(in[idx]);
}

// ---------------- RMSNorm over D=768, one block per token -------------------
__global__ void __launch_bounds__(256)
rmsnorm_k(const float* __restrict__ in, const float* __restrict__ w,
          bf16_t* __restrict__ out) {
    __shared__ float sm[256];
    const int n = blockIdx.x, tid = threadIdx.x;
    const float* row = in + (size_t)n * DD;
    float x0 = row[tid], x1 = row[tid + 256], x2 = row[tid + 512];
    sm[tid] = x0 * x0 + x1 * x1 + x2 * x2;
    __syncthreads();
    for (int o = 128; o > 0; o >>= 1) {
        if (tid < o) sm[tid] += sm[tid + o];
        __syncthreads();
    }
    float rms = rsqrtf(sm[0] * (1.0f / (float)DD) + 1e-6f);
    bf16_t* orow = out + (size_t)n * DD;
    orow[tid]       = f2bf(x0 * rms * w[tid]);
    orow[tid + 256] = f2bf(x1 * rms * w[tid + 256]);
    orow[tid + 512] = f2bf(x2 * rms * w[tid + 512]);
}

// ---------------- bf16 WMMA GEMM: C[M,Nc] = A[M,K] * Bt[Nc,K]^T, f32 out ----
// 8 waves/block; each wave owns a 32x64 tile (2 A-frags x 4 B-frags = 8 WMMA
// per K-step), software-pipelined so next-K fragment loads overlap WMMAs.
__global__ void __launch_bounds__(256)
wmma_gemm_f32(const bf16_t* __restrict__ A, const bf16_t* __restrict__ Bt,
              float* __restrict__ C, int M, int Nc, int K) {
    const int wid = threadIdx.x >> 5, lane = threadIdx.x & 31;
    const int gw = blockIdx.x * 8 + wid;
    const int tilesM = M >> 5, tilesN = Nc >> 6;
    if (gw >= tilesM * tilesN) return;
    const int mt = gw % tilesM, nt = gw / tilesM;
    const int row0 = mt << 5, col0 = nt << 6;
    const int lr = lane & 15, hiB = lane >> 4;

    const bf16_t* arow0 = A + (size_t)(row0 + lr) * K + hiB * 8;
    const bf16_t* arow1 = arow0 + (size_t)16 * K;
    const bf16_t* brow  = Bt + (size_t)(col0 + lr) * K + hiB * 8;

    v8f acc[2][4] = {};

    v16bf ca0 = load_afrag(arow0, 0);
    v16bf ca1 = load_afrag(arow1, 0);
    v16bf cb0 = load_afrag(brow, 0);
    v16bf cb1 = load_afrag(brow + (size_t)16 * K, 0);
    v16bf cb2 = load_afrag(brow + (size_t)32 * K, 0);
    v16bf cb3 = load_afrag(brow + (size_t)48 * K, 0);

#pragma unroll 2
    for (int k0 = 32; k0 < K; k0 += 32) {
        // issue next-iteration fragment loads first, then consume current
        v16bf na0 = load_afrag(arow0, k0);
        v16bf na1 = load_afrag(arow1, k0);
        v16bf nb0 = load_afrag(brow, k0);
        v16bf nb1 = load_afrag(brow + (size_t)16 * K, k0);
        v16bf nb2 = load_afrag(brow + (size_t)32 * K, k0);
        v16bf nb3 = load_afrag(brow + (size_t)48 * K, k0);
        acc[0][0] = wmma_bf16(ca0, cb0, acc[0][0]);
        acc[0][1] = wmma_bf16(ca0, cb1, acc[0][1]);
        acc[0][2] = wmma_bf16(ca0, cb2, acc[0][2]);
        acc[0][3] = wmma_bf16(ca0, cb3, acc[0][3]);
        acc[1][0] = wmma_bf16(ca1, cb0, acc[1][0]);
        acc[1][1] = wmma_bf16(ca1, cb1, acc[1][1]);
        acc[1][2] = wmma_bf16(ca1, cb2, acc[1][2]);
        acc[1][3] = wmma_bf16(ca1, cb3, acc[1][3]);
        ca0 = na0; ca1 = na1;
        cb0 = nb0; cb1 = nb1; cb2 = nb2; cb3 = nb3;
    }
    acc[0][0] = wmma_bf16(ca0, cb0, acc[0][0]);
    acc[0][1] = wmma_bf16(ca0, cb1, acc[0][1]);
    acc[0][2] = wmma_bf16(ca0, cb2, acc[0][2]);
    acc[0][3] = wmma_bf16(ca0, cb3, acc[0][3]);
    acc[1][0] = wmma_bf16(ca1, cb0, acc[1][0]);
    acc[1][1] = wmma_bf16(ca1, cb1, acc[1][1]);
    acc[1][2] = wmma_bf16(ca1, cb2, acc[1][2]);
    acc[1][3] = wmma_bf16(ca1, cb3, acc[1][3]);

    // C/D layout: vgpr r -> row = base + r + 8*(lane>=16), col = col0 + lane%16
#pragma unroll
    for (int mi = 0; mi < 2; ++mi) {
        const int rbase = row0 + mi * 16 + hiB * 8;
#pragma unroll
        for (int r = 0; r < 8; ++r) {
            float* cr = C + (size_t)(rbase + r) * Nc + col0 + lr;
            cr[0]  = acc[mi][0][r];
            cr[16] = acc[mi][1][r];
            cr[32] = acc[mi][2][r];
            cr[48] = acc[mi][3][r];
        }
    }
}

// ---------------- fused SwiGLU GEMM: H = bf16( silu(A*W1t^T) * (A*W3t^T) ) --
__global__ void __launch_bounds__(256)
wmma_gemm_swiglu(const bf16_t* __restrict__ A, const bf16_t* __restrict__ B1t,
                 const bf16_t* __restrict__ B3t, bf16_t* __restrict__ H,
                 int M, int Nc, int K) {
    const int wid = threadIdx.x >> 5, lane = threadIdx.x & 31;
    const int gw = blockIdx.x * 8 + wid;
    const int tilesM = M >> 4, tilesN = Nc >> 6;
    if (gw >= tilesM * tilesN) return;
    const int mt = gw % tilesM, nt = gw / tilesM;
    const int row0 = mt << 4, col0 = nt << 6;
    const int lr = lane & 15, hiB = lane >> 4;

    const bf16_t* arow  = A   + (size_t)(row0 + lr) * K + hiB * 8;
    const bf16_t* b1row = B1t + (size_t)(col0 + lr) * K + hiB * 8;
    const bf16_t* b3row = B3t + (size_t)(col0 + lr) * K + hiB * 8;

    v8f p[4] = {}, q[4] = {};
    for (int k0 = 0; k0 < K; k0 += 32) {
        v16bf a = load_afrag(arow, k0);
#pragma unroll
        for (int j = 0; j < 4; ++j) {
            v16bf b1 = load_afrag(b1row + (size_t)(16 * j) * K, k0);
            v16bf b3 = load_afrag(b3row + (size_t)(16 * j) * K, k0);
            p[j] = wmma_bf16(a, b1, p[j]);
            q[j] = wmma_bf16(a, b3, q[j]);
        }
    }
    const int rbase = row0 + hiB * 8;
#pragma unroll
    for (int r = 0; r < 8; ++r) {
        bf16_t* hr = H + (size_t)(rbase + r) * Nc + col0 + lr;
#pragma unroll
        for (int j = 0; j < 4; ++j)
            hr[16 * j] = f2bf(p[j][r] * sigmoidf_(p[j][r]) * q[j][r]);
    }
}

// ---------------- RG-LRU elementwise helper ---------------------------------
// proj layout per token: [u(384) | g(384) | v(384) | r(384) | i(384)] = 1920
__device__ __forceinline__ void lru_step(const float* p, int c, float mul,
                                         float& a, float& gated) {
    float v = p[2 * DC + c], r = p[3 * DC + c], ii = p[4 * DC + c];
    float la = mul * sigmoidf_(r);
    a = __expf(la);
    gated = sqrtf(fmaxf(1.0f - __expf(2.0f * la), 1e-6f)) * (sigmoidf_(ii) * v);
}

// Pass 1: per (b,c,chunk) compute (prod a, partial h) over the chunk.
__global__ void scan_part1_k(const float* __restrict__ proj,
                             const float* __restrict__ lam,
                             float* __restrict__ SA, float* __restrict__ SH) {
    int gid = blockIdx.x * blockDim.x + threadIdx.x;
    if (gid >= NB * DC * NCH) return;
    const int ch = gid % NCH, rem = gid / NCH;
    const int c = rem % DC, b = rem / DC;
    const float mul = -8.0f * log1pf(__expf(lam[c]));
    float Aacc = 1.0f, h = 0.0f;
    const int t0 = ch * CHUNK;
    for (int t = t0; t < t0 + CHUNK; ++t) {
        const float* p = proj + (size_t)(b * TT + t) * 1920;
        float a, gated;
        lru_step(p, c, mul, a, gated);
        Aacc *= a;
        h = a * h + gated;
    }
    SA[gid] = Aacc;
    SH[gid] = h;
}

// Pass 2: per (b,c) sequential combine of 16 chunk states -> h entering chunk.
__global__ void scan_part2_k(const float* __restrict__ SA,
                             const float* __restrict__ SH,
                             float* __restrict__ HST) {
    int gid = blockIdx.x * blockDim.x + threadIdx.x;
    if (gid >= NB * DC) return;
    const int base = gid * NCH;
    float run = 0.0f;
    for (int j = 0; j < NCH; ++j) {
        HST[base + j] = run;
        run = SA[base + j] * run + SH[base + j];
    }
}

// Pass 3: replay each chunk from its incoming state; fuse causal dwconv(K=3)
// * silu(g); write concat [conv_out | h] as bf16.
__global__ void scan_part3_k(const float* __restrict__ proj,
                             const float* __restrict__ convw,
                             const float* __restrict__ lam,
                             const float* __restrict__ HST,
                             bf16_t* __restrict__ mixcat) {
    int gid = blockIdx.x * blockDim.x + threadIdx.x;
    if (gid >= NB * DC * NCH) return;
    const int ch = gid % NCH, rem = gid / NCH;
    const int c = rem % DC, b = rem / DC;
    const float w0 = convw[c * 3 + 0], w1 = convw[c * 3 + 1], w2 = convw[c * 3 + 2];
    const float mul = -8.0f * log1pf(__expf(lam[c]));
    const int t0 = ch * CHUNK;
    const size_t rbase = (size_t)b * TT;
    float h   = HST[rem * NCH + ch];
    float um1 = (t0 >= 1) ? proj[(rbase + t0 - 1) * 1920 + c] : 0.0f;
    float um2 = (t0 >= 2) ? proj[(rbase + t0 - 2) * 1920 + c] : 0.0f;
    for (int t = t0; t < t0 + CHUNK; ++t) {
        const float* p = proj + (rbase + t) * 1920;
        float u = p[c], g = p[DC + c];
        float conv = u * w2 + um1 * w1 + um2 * w0;   // causal, XLA correlation
        um2 = um1; um1 = u;
        float co = conv * g * sigmoidf_(g);
        float a, gated;
        lru_step(p, c, mul, a, gated);
        h = a * h + gated;
        const size_t o = (rbase + t) * DD;
        mixcat[o + c]      = f2bf(co);
        mixcat[o + DC + c] = f2bf(h);
    }
}

// ---------------- momentum residual: vel = b*vel + mix; xmid = x + vel ------
__global__ void residual1_k(const float* __restrict__ x,
                            const float* __restrict__ vel_in,
                            const float* __restrict__ mixout,
                            const float* __restrict__ log_beta,
                            float* __restrict__ vel_out,
                            float* __restrict__ xmid) {
    int i = blockIdx.x * blockDim.x + threadIdx.x;
    if (i >= NDE) return;
    float beta = sigmoidf_(log_beta[0]);
    float v = beta * vel_in[i] + mixout[i];
    vel_out[i] = v;
    xmid[i] = x[i] + v;
}

__global__ void residual2_k(const float* __restrict__ xmid,
                            const float* __restrict__ ffnout,
                            float* __restrict__ xout) {
    int i = blockIdx.x * blockDim.x + threadIdx.x;
    if (i >= NDE) return;
    xout[i] = xmid[i] + ffnout[i];
}

// ---------------------------------------------------------------------------
extern "C" void kernel_launch(void* const* d_in, const int* in_sizes, int n_in,
                              void* d_out, int out_size, void* d_ws, size_t ws_size,
                              hipStream_t stream) {
    const float* x         = (const float*)d_in[0];
    const float* velocity  = (const float*)d_in[1];
    const float* pre_w     = (const float*)d_in[2];
    const float* conv_in_w = (const float*)d_in[3];
    const float* conv_dw_w = (const float*)d_in[4];
    const float* grif_v_w  = (const float*)d_in[5];
    const float* grif_r_w  = (const float*)d_in[6];
    const float* grif_i_w  = (const float*)d_in[7];
    const float* grif_lam  = (const float*)d_in[8];
    const float* out_w     = (const float*)d_in[9];
    const float* log_beta  = (const float*)d_in[10];
    const float* ffn_w     = (const float*)d_in[11];
    const float* w1        = (const float*)d_in[12];
    const float* w3        = (const float*)d_in[13];
    const float* w2        = (const float*)d_in[14];

    char* ws = (char*)d_ws;
    const size_t MB = 1024ull * 1024ull;
    bf16_t* WT_PROJ = (bf16_t*)(ws + 0 * MB);
    bf16_t* OUT_WT  = (bf16_t*)(ws + 4 * MB);
    bf16_t* W1T     = (bf16_t*)(ws + 8 * MB);
    bf16_t* W3T     = (bf16_t*)(ws + 12 * MB);
    bf16_t* W2T     = (bf16_t*)(ws + 16 * MB);
    bf16_t* XN      = (bf16_t*)(ws + 20 * MB);
    float*  PROJ    = (float*) (ws + 36 * MB);
    bf16_t* HBUF    = (bf16_t*)(ws + 36 * MB);
    bf16_t* MIXCAT  = (bf16_t*)(ws + 100 * MB);
    float*  MIXOUT  = (float*) (ws + 116 * MB);
    float*  FFNOUT  = (float*) (ws + 116 * MB);
    float*  XMID    = (float*) (ws + 144 * MB);
    float*  SA      = (float*) (ws + 170 * MB);
    float*  SH      = SA  + NB * DC * NCH;
    float*  HST     = SH  + NB * DC * NCH;

    float* out_x   = (float*)d_out;
    float* out_vel = (float*)d_out + NDE;

    auto cvt = [&](const float* in, int K, int Nc, bf16_t* out) {
        int total = K * Nc;
        convertT_k<<<(total + 255) / 256, 256, 0, stream>>>(in, K, Nc, out);
    };
    cvt(conv_in_w, DD, 2 * DC, WT_PROJ);
    cvt(grif_v_w,  DD, DG, WT_PROJ + (size_t)768  * DD);
    cvt(grif_r_w,  DD, DG, WT_PROJ + (size_t)1152 * DD);
    cvt(grif_i_w,  DD, DG, WT_PROJ + (size_t)1536 * DD);
    cvt(out_w,     DD, DD, OUT_WT);
    cvt(w1,        DD, DF, W1T);
    cvt(w3,        DD, DF, W3T);
    cvt(w2,        DF, DD, W2T);

    // 1) xn = rmsnorm(x) -> bf16
    rmsnorm_k<<<NTOK, 256, 0, stream>>>(x, pre_w, XN);

    // 2) proj = xn @ [conv_in|v|r|i] : [8192,768]x[768,1920]
    {
        int tiles = (NTOK / 32) * (1920 / 64);
        wmma_gemm_f32<<<(tiles + 7) / 8, 256, 0, stream>>>(XN, WT_PROJ, PROJ,
                                                           NTOK, 1920, DD);
    }
    // 3) chunked RG-LRU scan + conv gate -> mixcat bf16
    {
        int n1 = NB * DC * NCH;
        scan_part1_k<<<(n1 + 255) / 256, 256, 0, stream>>>(PROJ, grif_lam, SA, SH);
        scan_part2_k<<<(NB * DC + 255) / 256, 256, 0, stream>>>(SA, SH, HST);
        scan_part3_k<<<(n1 + 255) / 256, 256, 0, stream>>>(PROJ, conv_dw_w,
                                                           grif_lam, HST, MIXCAT);
    }
    // 4) mixout = mixcat @ out_w
    {
        int tiles = (NTOK / 32) * (DD / 64);
        wmma_gemm_f32<<<(tiles + 7) / 8, 256, 0, stream>>>(MIXCAT, OUT_WT,
                                                           MIXOUT, NTOK, DD, DD);
    }
    // 5) velocity/xmid
    residual1_k<<<(NDE + 255) / 256, 256, 0, stream>>>(x, velocity, MIXOUT,
                                                       log_beta, out_vel, XMID);
    // 6) normed2 = rmsnorm(xmid) -> bf16 (reuse XN)
    rmsnorm_k<<<NTOK, 256, 0, stream>>>(XMID, ffn_w, XN);

    // 7) h = silu(normed2@w1) * (normed2@w3) -> bf16
    {
        int tiles = (NTOK / 16) * (DF / 64);
        wmma_gemm_swiglu<<<(tiles + 7) / 8, 256, 0, stream>>>(XN, W1T, W3T,
                                                              HBUF, NTOK, DF, DD);
    }
    // 8) ffnout = h @ w2
    {
        int tiles = (NTOK / 32) * (DD / 64);
        wmma_gemm_f32<<<(tiles + 7) / 8, 256, 0, stream>>>(HBUF, W2T, FFNOUT,
                                                           NTOK, DD, DF);
    }
    // 9) x_out = xmid + ffnout
    residual2_k<<<(NDE + 255) / 256, 256, 0, stream>>>(XMID, FFNOUT, out_x);
}
